// FixableFeatureBlock_1529008358157
// MI455X (gfx1250) — matrix operand order repaired and checked by reading, not compiled, not run on hardware
//
#include <hip/hip_runtime.h>
#include <hip/hip_bf16.h>
#include <math.h>

// ---------------------------------------------------------------------------
// Shapes: B=1, CIN=128, COUT=64, H=W=32 (L=1024), K=3, HEADS=4, DH=8, INNER=32
// KW_CH = 64*128*9 = 73728, PRED_OUT = 73792
// ---------------------------------------------------------------------------

typedef __attribute__((ext_vector_type(16))) _Float16 v16h;
typedef __attribute__((ext_vector_type(8)))  float    v8f;

// ---- WMMA fragment loaders (CDNA5 16x16x32 f16 layouts, ISA 7.12.2) -------
// A (16x32 f16): lane m = lane%16; lanes<16 hold K {0..7,16..23}, lanes>=16
// hold K {8..15,24..31}; packed pairs per VGPR.
__device__ inline v16h load_a_frag_f32(const float* __restrict__ A, int row_base,
                                       int ld, int kk, int lane) {
  int hv = lane >> 4, m = lane & 15;
  const float* rowp = A + (size_t)(row_base + m) * ld;
  v16h a;
#pragma unroll
  for (int r = 0; r < 8; ++r) {
    int kb = kk + ((r < 4) ? (2 * r) : (16 + 2 * (r - 4))) + hv * 8;
    float2 p = *reinterpret_cast<const float2*>(rowp + kb);
    a[2 * r]     = (_Float16)p.x;
    a[2 * r + 1] = (_Float16)p.y;
  }
  return a;
}

// B (32x16 f16): lane n = lane%16; lanes<16 hold K {0..7,16..23}, lanes>=16 K {8..15,24..31}.
__device__ inline v16h load_b_frag_f32(const float* __restrict__ B, int n0,
                                       int ld, int kk, int lane) {
  int hv = lane >> 4, j = lane & 15;
  v16h b;
#pragma unroll
  for (int r = 0; r < 8; ++r) {
    int k = kk + ((r < 4) ? (2 * r) : (16 + 2 * (r - 4))) + hv * 8;
    b[2 * r]     = (_Float16)B[(size_t)k * ld + n0 + j];
    b[2 * r + 1] = (_Float16)B[(size_t)(k + 1) * ld + n0 + j];
  }
  return b;
}

__device__ inline v16h load_b_frag_lds(const _Float16* Bh /*[128][16]*/, int kk, int lane) {
  int hv = lane >> 4, j = lane & 15;
  v16h b;
#pragma unroll
  for (int r = 0; r < 8; ++r) {
    int k = kk + ((r < 4) ? (2 * r) : (16 + 2 * (r - 4))) + hv * 8;
    b[2 * r]     = Bh[k * 16 + j];
    b[2 * r + 1] = Bh[(k + 1) * 16 + j];
  }
  return b;
}

// ---------------------------------------------------------------------------
// Generic WMMA GEMM: C[M,N] = A[M,K] @ B[K,N] (+bias[M]) (+add[M,N])
// One wave per 16x16 C tile. M,N mult of 16; K mult of 32.
// HAS_BIAS/HAS_ADD templated so the epilogue is a clean store clause.
// ---------------------------------------------------------------------------
template <bool HAS_BIAS, bool HAS_ADD>
__global__ __launch_bounds__(32)
void wmma_gemm_f32(const float* __restrict__ A, const float* __restrict__ B,
                   const float* __restrict__ bias, const float* __restrict__ add,
                   float* __restrict__ C, int M, int N, int K) {
  int ntiles = N >> 4;
  int tn = blockIdx.x % ntiles, tm = blockIdx.x / ntiles;
  int lane = threadIdx.x & 31;
  int n0 = tn * 16, m0 = tm * 16;
  v8f c = {};
  for (int kk = 0; kk < K; kk += 32) {
    v16h a = load_a_frag_f32(A, m0, K, kk, lane);
    v16h b = load_b_frag_f32(B, n0, N, kk, lane);
    c = __builtin_amdgcn_wmma_f32_16x16x32_f16(false, a, false, b, (short)0, c,
                                               false, false);
  }
  int hv = lane >> 4, j = lane & 15;
#pragma unroll
  for (int r = 0; r < 8; ++r) {
    int m = m0 + r + 8 * hv;
    size_t idx = (size_t)m * N + n0 + j;
    float v = c[r];
    if (HAS_BIAS) v += bias[m];
    if (HAS_ADD) v += add[idx];
    C[idx] = v;
  }
}

// ---------------------------------------------------------------------------
// Fused dynamic convolution. One workgroup (8 waves) per 16-pixel tile.
// Never materializes dyn_w in HBM: the 1152x16 predicted-weight tile lives in
// LDS (72 KB, possible thanks to CDNA5's 320KB/WGP). Wp (37.8MB) stays hot in
// the 192MB L2 across the 64 workgroups. Neighborhood tile is staged with
// GLOBAL_LOAD_ASYNC_TO_LDS_B32 (ASYNCcnt) on gfx1250.
// ---------------------------------------------------------------------------
__global__ __launch_bounds__(256)
void dynconv_kernel(const float* __restrict__ x, const float* __restrict__ Wp,
                    const float* __restrict__ bp, float* __restrict__ y0) {
  extern __shared__ char smem[];
  float*    xnbr  = (float*)smem;                    // [128][3][18]  27648 B
  _Float16* xcolh = (_Float16*)(smem + 27648);       // [128][16]      4096 B
  float*    Gbuf  = (float*)(smem + 31744);          // [1152][16]    73728 B
  float*    inv1  = (float*)(smem + 105472);         // [9][16]         576 B
  float*    inv2  = inv1 + 144;                      // [128][16]      8192 B
  float*    partial = inv2 + 2048;                   // [256]          1024 B

  int tid = threadIdx.x;
  int blk = blockIdx.x;        // 64 blocks: (h, w-half)
  int h  = blk >> 1;
  int w0 = (blk & 1) << 4;
  int l0 = h * 32 + w0;

  // stage 3x18 neighborhood (zero-padded): async DMA to LDS for in-bounds
  // lanes, direct zero store for the padding lanes.
  for (int idx = tid; idx < 128 * 3 * 18; idx += 256) {
    int c = idx / 54, rem = idx % 54, dh = rem / 18, dw = rem % 18;
    int hh = h - 1 + dh, ww = w0 - 1 + dw;
    bool in = (hh >= 0 && hh < 32 && ww >= 0 && ww < 32);
#if defined(__gfx1250__)
    if (in) {
      unsigned lds_off = (unsigned)(size_t)(xnbr + idx);
      const float* gp = x + c * 1024 + hh * 32 + ww;
      asm volatile("global_load_async_to_lds_b32 %0, %1, off"
                   :: "v"(lds_off), "v"(gp) : "memory");
    } else {
      xnbr[idx] = 0.f;
    }
#else
    xnbr[idx] = in ? x[c * 1024 + hh * 32 + ww] : 0.f;
#endif
  }
  // center columns in f16 (needs conversion, so regular path)
  for (int idx = tid; idx < 128 * 16; idx += 256) {
    int c = idx >> 4, j = idx & 15;
    xcolh[idx] = (_Float16)x[c * 1024 + l0 + j];
  }
#if defined(__gfx1250__)
  asm volatile("s_wait_asynccnt 0" ::: "memory");
#endif
  __syncthreads();

  int lane = tid & 31, wave = tid >> 5;
  int hv = lane >> 4, lm = lane & 15;

  for (int o = 0; o < 64; ++o) {
    const float* Wpo = Wp + (size_t)o * 1152 * 128;
    // G(1152x16) = Wp_o(1152x128) @ xcol(128x16) via WMMA, tiles split on waves
    for (int t = wave; t < 72; t += 8) {
      int r0 = t * 16;
      __builtin_prefetch(Wpo + (size_t)(r0 + 128) * 128, 0, 0);
      v8f acc = {};
#pragma unroll
      for (int kk = 0; kk < 128; kk += 32) {
        v16h a = load_a_frag_f32(Wpo, r0, 128, kk, lane);
        v16h b = load_b_frag_lds(xcolh, kk, lane);
        acc = __builtin_amdgcn_wmma_f32_16x16x32_f16(false, a, false, b,
                                                     (short)0, acc, false, false);
      }
#pragma unroll
      for (int r = 0; r < 8; ++r)
        Gbuf[(r0 + r + 8 * hv) * 16 + lm] = acc[r];
    }
    __syncthreads();

    // l2norm over c (axis=2): one (k,l) pair per thread
    if (tid < 144) {
      int k = tid / 16, l = tid & 15;
      float s = 0.f;
      for (int c = 0; c < 128; ++c) {
        float v = Gbuf[(c * 9 + k) * 16 + l];
        s += v * v;
      }
      inv1[tid] = 1.f / fmaxf(sqrtf(s), 1e-12f);
    }
    __syncthreads();

    // l2norm over k (axis=3) of the already-c-normalized weights
    for (int p = tid; p < 2048; p += 256) {
      int c = p >> 4, l = p & 15;
      float s = 0.f;
#pragma unroll
      for (int k = 0; k < 9; ++k) {
        float v = Gbuf[(c * 9 + k) * 16 + l] * inv1[k * 16 + l];
        s += v * v;
      }
      inv2[p] = 1.f / fmaxf(sqrtf(s), 1e-12f);
    }
    __syncthreads();

    // contraction with patches: y[o,l] = sum_{c,k} w_norm * patch
    {
      int part = tid >> 4, l = tid & 15;
      float s = 0.f;
      for (int c = part * 8; c < part * 8 + 8; ++c) {
        float i2 = inv2[c * 16 + l];
#pragma unroll
        for (int k = 0; k < 9; ++k) {
          float w = Gbuf[(c * 9 + k) * 16 + l] * inv1[k * 16 + l] * i2;
          s += w * xnbr[(c * 3 + k / 3) * 18 + l + (k % 3)];
        }
      }
      partial[tid] = s;
    }
    __syncthreads();
    if (tid < 16) {
      int l = tid;
      float tot = 0.f;
#pragma unroll
      for (int p = 0; p < 16; ++p) tot += partial[p * 16 + l];
      // dynamic bias: pred row KW_CH + o
      const float* wb = Wp + (size_t)(73728 + o) * 128;
      float db = bp[73728 + o];
      for (int c = 0; c < 128; ++c) db += wb[c] * xnbr[(c * 3 + 1) * 18 + l + 1];
      y0[o * 1024 + l0 + l] = tot + db;
    }
    __syncthreads();  // Gbuf reused next o
  }
}

// ---------------------------------------------------------------------------
// Per-channel batchnorm stats over L=1024
// ---------------------------------------------------------------------------
__global__ __launch_bounds__(256)
void bnstats_kernel(const float* __restrict__ src, float* __restrict__ mean,
                    float* __restrict__ var, int N) {
  __shared__ float s1[256], s2[256];
  int c = blockIdx.x, tid = threadIdx.x;
  float a = 0.f, b = 0.f;
  for (int i = tid; i < N; i += 256) {
    float v = src[c * N + i];
    a += v;
    b += v * v;
  }
  s1[tid] = a; s2[tid] = b;
  __syncthreads();
  for (int s = 128; s > 0; s >>= 1) {
    if (tid < s) { s1[tid] += s1[tid + s]; s2[tid] += s2[tid + s]; }
    __syncthreads();
  }
  if (tid == 0) {
    float m = s1[0] / N;
    mean[c] = m;
    var[c] = s2[0] / N - m * m;
  }
}

__global__ void bn_gelu_kernel(const float* __restrict__ src,
                               const float* __restrict__ mean,
                               const float* __restrict__ var,
                               const float* __restrict__ g,
                               const float* __restrict__ b,
                               float* __restrict__ out, int N, int total) {
  int i = blockIdx.x * blockDim.x + threadIdx.x;
  if (i >= total) return;
  int c = i / N;
  float xh = (src[i] - mean[c]) * rsqrtf(var[c] + 1e-5f);
  float y = xh * g[c] + b[c];
  out[i] = 0.5f * y * (1.f + erff(y * 0.70710678118f));
}

// bilinear resize pos_bias (4,16,16) -> (4,32,32), half-pixel centers
__global__ void pos_resize_kernel(const float* __restrict__ pb, float* __restrict__ pos) {
  int i = blockIdx.x * blockDim.x + threadIdx.x;
  if (i >= 4096) return;
  int hd = i >> 10, p = i & 1023;
  int r = p >> 5, cl = p & 31;
  float fr = (r + 0.5f) * 0.5f - 0.5f;
  float fc = (cl + 0.5f) * 0.5f - 0.5f;
  int r0 = (int)floorf(fr), c0 = (int)floorf(fc);
  float ar = fr - r0, ac = fc - c0;
  int r0c = min(max(r0, 0), 15), r1c = min(max(r0 + 1, 0), 15);
  int c0c = min(max(c0, 0), 15), c1c = min(max(c0 + 1, 0), 15);
  const float* base = pb + hd * 256;
  float v = (1 - ar) * ((1 - ac) * base[r0c * 16 + c0c] + ac * base[r0c * 16 + c1c]) +
            ar * ((1 - ac) * base[r1c * 16 + c0c] + ac * base[r1c * 16 + c1c]);
  pos[i] = v;
}

// ---------------------------------------------------------------------------
// Flash-style attention: one block per (head, query i). DH=8, L=1024.
// Scores stay in LDS; no LxL matrix in HBM.
// qkv rows: s*32 + h*8 + d   (s=0:q, 1:k, 2:v)
// ---------------------------------------------------------------------------
__global__ __launch_bounds__(256)
void attention_kernel(const float* __restrict__ qkv, const float* __restrict__ pos,
                      float* __restrict__ abuf) {
  __shared__ float sc[1024];
  __shared__ float red[256];
  int h = blockIdx.x >> 10, i = blockIdx.x & 1023, tid = threadIdx.x;
  const float scale = 0.35355339059f;  // 1/sqrt(8)
  float qi[8];
#pragma unroll
  for (int d = 0; d < 8; ++d) qi[d] = qkv[(h * 8 + d) * 1024 + i] * scale;
  float pi = pos[h * 1024 + i];
  float lmax = -1e30f;
  for (int j = tid; j < 1024; j += 256) {
    float s = 0.f;
#pragma unroll
    for (int d = 0; d < 8; ++d) s += qi[d] * qkv[(32 + h * 8 + d) * 1024 + j];
    s += pi + pos[h * 1024 + j];
    sc[j] = s;
    lmax = fmaxf(lmax, s);
  }
  red[tid] = lmax;
  __syncthreads();
  for (int s = 128; s > 0; s >>= 1) {
    if (tid < s) red[tid] = fmaxf(red[tid], red[tid + s]);
    __syncthreads();
  }
  float m = red[0];
  __syncthreads();
  float lsum = 0.f;
  for (int j = tid; j < 1024; j += 256) {
    float e = __expf(sc[j] - m);
    sc[j] = e;
    lsum += e;
  }
  red[tid] = lsum;
  __syncthreads();
  for (int s = 128; s > 0; s >>= 1) {
    if (tid < s) red[tid] += red[tid + s];
    __syncthreads();
  }
  float inv = 1.f / red[0];
  __syncthreads();
#pragma unroll
  for (int d = 0; d < 8; ++d) {
    float acc = 0.f;
    for (int j = tid; j < 1024; j += 256) acc += sc[j] * qkv[(64 + h * 8 + d) * 1024 + j];
    red[tid] = acc;
    __syncthreads();
    for (int s = 128; s > 0; s >>= 1) {
      if (tid < s) red[tid] += red[tid + s];
      __syncthreads();
    }
    if (tid == 0) abuf[(h * 8 + d) * 1024 + i] = red[0] * inv;
    __syncthreads();
  }
}

// per-pixel layernorm over 64 channels
__global__ __launch_bounds__(64)
void ln_pixel_kernel(const float* __restrict__ z, const float* __restrict__ g,
                     const float* __restrict__ b, float* __restrict__ out) {
  __shared__ float s1[64], s2[64];
  int l = blockIdx.x, c = threadIdx.x;
  float v = z[c * 1024 + l];
  s1[c] = v; s2[c] = v * v;
  __syncthreads();
  for (int s = 32; s > 0; s >>= 1) {
    if (c < s) { s1[c] += s1[c + s]; s2[c] += s2[c + s]; }
    __syncthreads();
  }
  float mu = s1[0] / 64.f, var = s2[0] / 64.f - mu * mu;
  out[c * 1024 + l] = (v - mu) * rsqrtf(var + 1e-5f) * g[c] + b[c];
}

// ---------------------------------------------------------------------------
extern "C" void kernel_launch(void* const* d_in, const int* in_sizes, int n_in,
                              void* d_out, int out_size, void* d_ws, size_t ws_size,
                              hipStream_t stream) {
  const float* x        = (const float*)d_in[0];
  const float* Wp       = (const float*)d_in[1];
  const float* bp       = (const float*)d_in[2];
  const float* g1       = (const float*)d_in[3];
  const float* b1       = (const float*)d_in[4];
  const float* Wqkv     = (const float*)d_in[5];
  const float* Wpa      = (const float*)d_in[6];
  const float* bpa      = (const float*)d_in[7];
  const float* pos_bias = (const float*)d_in[8];
  const float* g_ln     = (const float*)d_in[9];
  const float* b_ln     = (const float*)d_in[10];
  const float* g2       = (const float*)d_in[11];
  const float* b2       = (const float*)d_in[12];
  const float* Wproj    = (const float*)d_in[13];
  const float* bproj    = (const float*)d_in[14];
  const float* Wsc      = (const float*)d_in[15];
  const float* bsc      = (const float*)d_in[16];
  float* out = (float*)d_out;

  float* ws    = (float*)d_ws;
  float* resid = ws;                 // 64x1024
  float* y0    = ws + 65536;         // 64x1024
  float* yact  = ws + 131072;        // 64x1024
  float* qkv   = ws + 196608;        // 96x1024
  float* abuf  = ws + 294912;        // 32x1024
  float* z     = ws + 327680;        // 64x1024
  float* zln   = ws + 393216;        // 64x1024
  float* act2  = ws + 458752;        // 64x1024
  float* pos   = ws + 524288;        // 4x1024
  float* mean1 = ws + 528384;
  float* var1  = mean1 + 64;
  float* mean2 = var1 + 64;
  float* var2  = mean2 + 64;

  // pos bias resize (independent)
  pos_resize_kernel<<<16, 256, 0, stream>>>(pos_bias, pos);
  // residual path: resid = Wsc @ x + bsc        (64x1024x128)
  wmma_gemm_f32<true, false><<<256, 32, 0, stream>>>(Wsc, x, bsc, nullptr, resid,
                                                     64, 1024, 128);
  // fused dynamic conv (dominant cost): y0
  dynconv_kernel<<<64, 256, 115264, stream>>>(x, Wp, bp, y0);
  // BN1 + gelu
  bnstats_kernel<<<64, 256, 0, stream>>>(y0, mean1, var1, 1024);
  bn_gelu_kernel<<<256, 256, 0, stream>>>(y0, mean1, var1, g1, b1, yact, 1024, 65536);
  // qkv = Wqkv @ yact                            (96x1024x64)
  wmma_gemm_f32<false, false><<<384, 32, 0, stream>>>(Wqkv, yact, nullptr, nullptr,
                                                      qkv, 96, 1024, 64);
  // attention -> abuf (32x1024)
  attention_kernel<<<4096, 256, 0, stream>>>(qkv, pos, abuf);
  // z = Wpa @ abuf + bpa + yact                  (64x1024x32)
  wmma_gemm_f32<true, true><<<256, 32, 0, stream>>>(Wpa, abuf, bpa, yact, z,
                                                    64, 1024, 32);
  // per-pixel layernorm
  ln_pixel_kernel<<<1024, 64, 0, stream>>>(z, g_ln, b_ln, zln);
  // BN2 + gelu
  bnstats_kernel<<<64, 256, 0, stream>>>(zln, mean2, var2, 1024);
  bn_gelu_kernel<<<256, 256, 0, stream>>>(zln, mean2, var2, g2, b2, act2, 1024, 65536);
  // out = Wproj @ act2 + bproj + resid           (64x1024x64)
  wmma_gemm_f32<true, true><<<256, 32, 0, stream>>>(Wproj, act2, bproj, resid, out,
                                                    64, 1024, 64);
}
